// GNN_net_77094662963450
// MI455X (gfx1250) — compile-verified, matrix operand order checked by reading, not run on hardware
//
#include <hip/hip_runtime.h>
#include <hip/hip_bf16.h>

// ---------------------------------------------------------------------------
// GNN forward for CDNA5 (gfx1250): every dense layer runs on
// v_wmma_f32_16x16x32_bf16. One wave32 per block owns a 16-row tile.
// LDS buffers are lifetime-overlaid (28 KB/block -> ~11 workgroups/WGP).
// All activation traffic is 16B-vectorized (b128 loads, packed bf16 stores).
// ---------------------------------------------------------------------------

typedef __bf16 bf16_t;
typedef __attribute__((ext_vector_type(4)))  bf16_t v4bf;
typedef __attribute__((ext_vector_type(8)))  bf16_t v8bf;
typedef __attribute__((ext_vector_type(16))) bf16_t v16bf;
typedef __attribute__((ext_vector_type(8)))  float  v8f;

constexpr int DV = 128;   // embedding dim
constexpr int DH = 256;   // hidden dim
constexpr float LN_EPS = 1e-5f;

struct MlpDev {
  const bf16_t* w1; const float* b1; const float* g1; const float* be1;
  const bf16_t* w2; const float* b2; const float* g2; const float* be2;
};

// Shared memory, 28 KB/block with lifetime overlays:
//   bytes [0,12288)      : Xb 16x384 bf16 (MLP input); after layer1 the same
//                          region is Hb 16x256 bf16 (post-LN activations)
//   bytes [12288,28672)  : Hf 16x256 f32 (pre-LN); layer2 uses only the first
//                          8 KB, so Of 16x128 f32 lives in the second 8 KB
#define SMEM_DECL                                            \
  __shared__ alignas(16) char smem[12288 + 16384];           \
  bf16_t* Xb = (bf16_t*)smem;                                \
  bf16_t* Hb = (bf16_t*)smem;                                \
  float*  Hf = (float*)(smem + 12288);                       \
  float*  Of = (float*)(smem + 12288 + 8192);

// Load a 16x32 bf16 WMMA operand fragment. Lane L (row/col = L&15, half
// h = L>>4) holds K = h*8..h*8+7 in v[0..7] and K = 16+h*8.. in v[8..15].
__device__ __forceinline__ v16bf ld_frag(const bf16_t* p) {
  v8bf lo = *(const v8bf*)(p);
  v8bf hi = *(const v8bf*)(p + 16);
  return __builtin_shufflevector(lo, hi, 0,1,2,3,4,5,6,7,8,9,10,11,12,13,14,15);
}

// Y(pre-LN) = X[16 x DIN] @ W[DOUT x DIN]^T + bias, via WMMA; raw result to Hf.
template<int DIN, int DOUT>
__device__ __forceinline__ void wave_dense(const bf16_t* __restrict__ X,
                                           const bf16_t* __restrict__ W,
                                           const float*  __restrict__ bias,
                                           float* __restrict__ Hfout) {
  const int lane = threadIdx.x & 31;
  const int r = lane & 15;      // A row / B column / D column
  const int h = lane >> 4;
  for (int nt = 0; nt < DOUT / 16; ++nt) {
    v8f acc = {};
    const bf16_t* wrow = W + (size_t)(nt * 16 + r) * DIN + h * 8;
    const bf16_t* xrow = X + r * DIN + h * 8;
#pragma unroll
    for (int kt = 0; kt < DIN / 32; ++kt) {
      v16bf a = ld_frag(xrow + kt * 32);
      v16bf b = ld_frag(wrow + kt * 32);
      acc = __builtin_amdgcn_wmma_f32_16x16x32_bf16(
          /*neg_a=*/false, a, /*neg_b=*/false, b,
          /*c_mod=*/(short)0, acc, /*reuse_a=*/false, /*reuse_b=*/false);
    }
    const float bn = bias[nt * 16 + r];
#pragma unroll
    for (int j = 0; j < 8; ++j)   // D element (m = j + 8*h, n = nt*16 + r)
      Hfout[(j + 8 * h) * DOUT + nt * 16 + r] = acc[j] + bn;
  }
}

// Per-row LayerNorm + ReLU over DOUT. Two lanes per row (half each) with a
// cross-half shuffle combine. Fully float4/v4bf vectorized.
template<int DOUT>
__device__ __forceinline__ void wave_ln_relu(const float* __restrict__ Hfin,
                                             const float* __restrict__ g,
                                             const float* __restrict__ be,
                                             bf16_t* bf_out, int bstride,
                                             float* f_out, int fstride,
                                             int rows) {
  const int lane = threadIdx.x & 31;
  const int r = lane & 15;
  const int h = lane >> 4;
  constexpr int HALF = DOUT / 2;
  const float* row = Hfin + r * DOUT + h * HALF;
  float s = 0.f, s2 = 0.f;
#pragma unroll
  for (int i = 0; i < HALF / 4; ++i) {
    float4 v = *(const float4*)(row + i * 4);
    s  += (v.x + v.y) + (v.z + v.w);
    s2 += (v.x * v.x + v.y * v.y) + (v.z * v.z + v.w * v.w);
  }
  s  += __shfl_xor(s, 16, 32);
  s2 += __shfl_xor(s2, 16, 32);
  const float m   = s / (float)DOUT;
  const float var = s2 / (float)DOUT - m * m;
  const float inv = rsqrtf(var + LN_EPS);
#pragma unroll
  for (int i = 0; i < HALF / 4; ++i) {
    const int c = h * HALF + i * 4;
    float4 v  = *(const float4*)(row + i * 4);
    float4 gv = *(const float4*)(g + c);
    float4 bv = *(const float4*)(be + c);
    float o0 = fmaxf((v.x - m) * inv * gv.x + bv.x, 0.f);
    float o1 = fmaxf((v.y - m) * inv * gv.y + bv.y, 0.f);
    float o2 = fmaxf((v.z - m) * inv * gv.z + bv.z, 0.f);
    float o3 = fmaxf((v.w - m) * inv * gv.w + bv.w, 0.f);
    if (r < rows) {
      if (bf_out) {
        v4bf bb = { (bf16_t)o0, (bf16_t)o1, (bf16_t)o2, (bf16_t)o3 };
        *(v4bf*)(bf_out + (size_t)r * bstride + c) = bb;
      }
      if (f_out)
        *(float4*)(f_out + (size_t)r * fstride + c) = make_float4(o0, o1, o2, o3);
    }
  }
}

// Full 2-layer MLP (DIN -> 256 -> 128), Linear+LN+ReLU twice.
// NOTE: Hb overlays Xb (X is dead after dense1); Of overlays Hf[8K:16K].
template<int DIN>
__device__ __forceinline__ void wave_mlp(const bf16_t* Xb, float* Hf, bf16_t* Hb,
                                         const MlpDev& P,
                                         bf16_t* bf_out, int bstride,
                                         float* f_out, int fstride, int rows) {
  wave_dense<DIN, DH>(Xb, P.w1, P.b1, Hf);
  __syncthreads();
  wave_ln_relu<DH>(Hf, P.g1, P.be1, Hb, DH, nullptr, 0, 16);
  __syncthreads();
  wave_dense<DH, DV>(Hb, P.w2, P.b2, Hf);   // writes Hf[16][128] (first 8 KB)
  __syncthreads();
  wave_ln_relu<DV>(Hf, P.g2, P.be2, bf_out, bstride, f_out, fstride, rows);
}

// ---------------------------------------------------------------------------
// Utility kernels
// ---------------------------------------------------------------------------
__global__ void k_cvt_bf16(const float* __restrict__ src, bf16_t* __restrict__ dst, int n) {
  for (int i = blockIdx.x * blockDim.x + threadIdx.x; i < n; i += gridDim.x * blockDim.x)
    dst[i] = (bf16_t)src[i];
}

__global__ void k_zero_f32(float* __restrict__ p, int n) {
  for (int i = blockIdx.x * blockDim.x + threadIdx.x; i < n; i += gridDim.x * blockDim.x)
    p[i] = 0.f;
}

__global__ void k_degree(const int* __restrict__ idx, float* __restrict__ deg, int n) {
  for (int i = blockIdx.x * blockDim.x + threadIdx.x; i < n; i += gridDim.x * blockDim.x)
    atomicAdd(&deg[idx[i]], 1.0f);
}

// ---------------------------------------------------------------------------
// hidden = MLP_v(batch_token)   [N,128] -> fp32 + bf16 copies
// ---------------------------------------------------------------------------
__global__ void __launch_bounds__(32)
k_node_mlp(const float* __restrict__ x, int N, MlpDev P,
           float* __restrict__ hf, bf16_t* __restrict__ hb) {
  SMEM_DECL;
  const int lane = threadIdx.x & 31;
  const int n0 = blockIdx.x * 16;
  const int rows = (N - n0 < 16) ? (N - n0) : 16;
  for (int i = lane; i < 16 * (DV / 4); i += 32) {
    const int r = i / (DV / 4);
    const int c = (i % (DV / 4)) * 4;
    float4 v = make_float4(0.f, 0.f, 0.f, 0.f);
    if (r < rows) v = *(const float4*)(x + (size_t)(n0 + r) * DV + c);
    v4bf bb = { (bf16_t)v.x, (bf16_t)v.y, (bf16_t)v.z, (bf16_t)v.w };
    *(v4bf*)(Xb + r * DV + c) = bb;
  }
  __syncthreads();
  wave_mlp<DV>(Xb, Hf, Hb, P, hb + (size_t)n0 * DV, DV, hf + (size_t)n0 * DV, DV, rows);
  (void)Of;
}

// ---------------------------------------------------------------------------
// edge embedding: MLP_e(indicate[:,None]); layer1 is rank-1 (d_in = 1).
// ---------------------------------------------------------------------------
__global__ void __launch_bounds__(32)
k_edge_emb(const float* __restrict__ ind, int E,
           const float* __restrict__ w1, const float* __restrict__ b1,
           MlpDev P, bf16_t* __restrict__ emb) {
  SMEM_DECL;
  (void)Xb; (void)Of;
  const int lane = threadIdx.x & 31;
  const int e0 = blockIdx.x * 16;
  const int rows = (E - e0 < 16) ? (E - e0) : 16;
  const int r = lane & 15, h = lane >> 4;
  const float xr = (r < rows) ? ind[e0 + r] : 0.f;
#pragma unroll
  for (int i = 0; i < (DH / 2) / 4; ++i) {
    const int j = h * (DH / 2) + i * 4;
    float4 w = *(const float4*)(w1 + j);
    float4 b = *(const float4*)(b1 + j);
    *(float4*)(Hf + r * DH + j) =
        make_float4(w.x * xr + b.x, w.y * xr + b.y, w.z * xr + b.z, w.w * xr + b.w);
  }
  __syncthreads();
  wave_ln_relu<DH>(Hf, P.g1, P.be1, Hb, DH, nullptr, 0, 16);
  __syncthreads();
  wave_dense<DH, DV>(Hb, P.w2, P.b2, Hf);
  __syncthreads();
  wave_ln_relu<DV>(Hf, P.g2, P.be2, emb + (size_t)e0 * DV, DV, nullptr, 0, rows);
}

// ---------------------------------------------------------------------------
// per-edge message MLP: s = MLP(concat([h[other], h[self], edge_emb]))
// then scatter-sum into sums[self] with f32 atomics.
// ---------------------------------------------------------------------------
__global__ void __launch_bounds__(32)
k_edge_msg(const bf16_t* __restrict__ hbf, const bf16_t* __restrict__ emb,
           const int* __restrict__ idx_self, const int* __restrict__ idx_other,
           int E, MlpDev P, float* __restrict__ sums) {
  SMEM_DECL;
  const int lane = threadIdx.x & 31;
  const int e0 = blockIdx.x * 16;
  const int rows = (E - e0 < 16) ? (E - e0) : 16;
  constexpr int DIN = 3 * DV;
  for (int i = lane; i < 16 * (DIN / 8); i += 32) {   // 16B chunks
    const int r = i / (DIN / 8);
    const int c = (i % (DIN / 8)) * 8;
    uint4 v = make_uint4(0u, 0u, 0u, 0u);
    if (r < rows) {
      const int e = e0 + r;
      const bf16_t* src;
      if (c < DV)            src = hbf + (size_t)idx_other[e] * DV + c;
      else if (c < 2 * DV)   src = hbf + (size_t)idx_self[e] * DV + (c - DV);
      else                   src = emb + (size_t)e * DV + (c - 2 * DV);
      v = *(const uint4*)src;
    }
    *(uint4*)(Xb + r * DIN + c) = v;
  }
  __syncthreads();
  wave_mlp<DIN>(Xb, Hf, Hb, P, nullptr, 0, Of, DV, 16);
  __syncthreads();
  for (int i = lane; i < 16 * (DV / 4); i += 32) {
    const int r = i / (DV / 4);
    const int c = (i % (DV / 4)) * 4;
    if (r < rows) {
      float4 v = *(const float4*)(Of + r * DV + c);
      float* dst = sums + (size_t)idx_self[e0 + r] * DV + c;
      atomicAdd(dst + 0, v.x); atomicAdd(dst + 1, v.y);
      atomicAdd(dst + 2, v.z); atomicAdd(dst + 3, v.w);
    }
  }
}

// ---------------------------------------------------------------------------
// node update: s = sums/max(deg,1) + mask*token; x = [h, s_p, s_c];
// hidden = relu(hidden + MLP_aggr(x)); refresh bf16 copy in place.
// ---------------------------------------------------------------------------
__global__ void __launch_bounds__(32)
k_node_aggr(const float* __restrict__ hin, bf16_t* hbf,
            const float* __restrict__ sums_p, const float* __restrict__ sums_c,
            const float* __restrict__ deg_p, const float* __restrict__ deg_c,
            const float* __restrict__ pm, const float* __restrict__ cm,
            const float* __restrict__ stok, const float* __restrict__ etok,
            int N, MlpDev P, float* __restrict__ hout) {
  SMEM_DECL;
  const int lane = threadIdx.x & 31;
  const int n0 = blockIdx.x * 16;
  const int rows = (N - n0 < 16) ? (N - n0) : 16;
  constexpr int DIN = 3 * DV;
  // slot 0: current hidden (bf16, 16B chunks)
  for (int i = lane; i < 16 * (DV / 8); i += 32) {
    const int r = i / (DV / 8);
    const int c = (i % (DV / 8)) * 8;
    uint4 v = make_uint4(0u, 0u, 0u, 0u);
    if (r < rows) v = *(const uint4*)(hbf + (size_t)(n0 + r) * DV + c);
    *(uint4*)(Xb + r * DIN + c) = v;
  }
  // slots 1,2: scatter-mean + mask*token (fp32 math, quantize to bf16)
  for (int i = lane; i < 16 * (2 * DV / 4); i += 32) {
    const int r  = i / (2 * DV / 4);
    const int cc = (i % (2 * DV / 4)) * 4;
    float4 v = make_float4(0.f, 0.f, 0.f, 0.f);
    if (r < rows) {
      const int vtx = n0 + r;
      if (cc < DV) {
        const float sc = 1.f / fmaxf(deg_p[vtx], 1.f);
        const float mk = pm[vtx];
        float4 sv = *(const float4*)(sums_p + (size_t)vtx * DV + cc);
        float4 tv = *(const float4*)(stok + cc);
        v = make_float4(sv.x * sc + mk * tv.x, sv.y * sc + mk * tv.y,
                        sv.z * sc + mk * tv.z, sv.w * sc + mk * tv.w);
      } else {
        const int c2 = cc - DV;
        const float sc = 1.f / fmaxf(deg_c[vtx], 1.f);
        const float mk = cm[vtx];
        float4 sv = *(const float4*)(sums_c + (size_t)vtx * DV + c2);
        float4 tv = *(const float4*)(etok + c2);
        v = make_float4(sv.x * sc + mk * tv.x, sv.y * sc + mk * tv.y,
                        sv.z * sc + mk * tv.z, sv.w * sc + mk * tv.w);
      }
    }
    v4bf bb = { (bf16_t)v.x, (bf16_t)v.y, (bf16_t)v.z, (bf16_t)v.w };
    *(v4bf*)(Xb + r * DIN + DV + cc) = bb;
  }
  __syncthreads();
  wave_mlp<DIN>(Xb, Hf, Hb, P, nullptr, 0, Of, DV, 16);
  __syncthreads();
  for (int i = lane; i < 16 * (DV / 4); i += 32) {
    const int r = i / (DV / 4);
    const int c = (i % (DV / 4)) * 4;
    if (r < rows) {
      const size_t o = (size_t)(n0 + r) * DV + c;
      float4 hv = *(const float4*)(hin + o);
      float4 ov = *(const float4*)(Of + r * DV + c);
      float4 res = make_float4(fmaxf(hv.x + ov.x, 0.f), fmaxf(hv.y + ov.y, 0.f),
                               fmaxf(hv.z + ov.z, 0.f), fmaxf(hv.w + ov.w, 0.f));
      *(float4*)(hout + o) = res;
      v4bf bb = { (bf16_t)res.x, (bf16_t)res.y, (bf16_t)res.z, (bf16_t)res.w };
      *(v4bf*)(hbf + o) = bb;
    }
  }
}

// ---------------------------------------------------------------------------
// Host launcher
// ---------------------------------------------------------------------------
extern "C" void kernel_launch(void* const* d_in, const int* in_sizes, int n_in,
                              void* d_out, int out_size, void* d_ws, size_t ws_size,
                              hipStream_t stream) {
  (void)n_in; (void)out_size; (void)ws_size;
  const float* batch  = (const float*)d_in[0];
  const int*   p_node = (const int*)d_in[1];
  const int*   c_node = (const int*)d_in[2];
  const float* p_ind  = (const float*)d_in[3];
  const float* c_ind  = (const float*)d_in[4];
  const float* pm     = (const float*)d_in[5];
  const float* cm     = (const float*)d_in[6];
  const float* stok   = (const float*)d_in[7];
  const float* etok   = (const float*)d_in[8];
  const int N = in_sizes[0] / DV;
  const int E = in_sizes[1];

  struct HostMlp { const float *w1, *b1, *g1, *be1, *w2, *b2, *g2, *be2; };
  auto get_mlp = [&](int base) -> HostMlp {
    const float* p[8];
    for (int i = 0; i < 8; ++i) p[i] = (const float*)d_in[base + i];
    HostMlp m;
    if (in_sizes[base + 4] == DV * DH) {
      // dict insertion order: w1,b1,g1,be1,w2,b2,g2,be2
      m.w1 = p[0]; m.b1 = p[1]; m.g1 = p[2]; m.be1 = p[3];
      m.w2 = p[4]; m.b2 = p[5]; m.g2 = p[6]; m.be2 = p[7];
    } else {
      // jax tree (alphabetical): b1,b2,be1,be2,g1,g2,w1,w2
      m.b1 = p[0]; m.b2 = p[1]; m.be1 = p[2]; m.be2 = p[3];
      m.g1 = p[4]; m.g2 = p[5]; m.w1  = p[6]; m.w2  = p[7];
    }
    return m;
  };
  HostMlp Pv = get_mlp(9), Pe = get_mlp(17), Pp = get_mlp(25),
          Pc = get_mlp(33), Pa = get_mlp(41);
  // d_in[49] = num_hops (device scalar, unreadable under graph capture): 3.

  // ---- workspace carve (256B aligned) ----
  char* wsb = (char*)d_ws;
  size_t off = 0;
  auto carve = [&](size_t bytes) -> void* {
    off = (off + 255) & ~(size_t)255;
    void* p = wsb + off; off += bytes; return p;
  };
  bf16_t* v_w1 = (bf16_t*)carve((size_t)DH * DV * 2);
  bf16_t* v_w2 = (bf16_t*)carve((size_t)DV * DH * 2);
  bf16_t* e_w2 = (bf16_t*)carve((size_t)DV * DH * 2);
  bf16_t* p_w1 = (bf16_t*)carve((size_t)DH * 3 * DV * 2);
  bf16_t* p_w2 = (bf16_t*)carve((size_t)DV * DH * 2);
  bf16_t* c_w1 = (bf16_t*)carve((size_t)DH * 3 * DV * 2);
  bf16_t* c_w2 = (bf16_t*)carve((size_t)DV * DH * 2);
  bf16_t* a_w1 = (bf16_t*)carve((size_t)DH * 3 * DV * 2);
  bf16_t* a_w2 = (bf16_t*)carve((size_t)DV * DH * 2);
  float*  hidden = (float*)carve((size_t)N * DV * 4);
  bf16_t* hbf    = (bf16_t*)carve((size_t)N * DV * 2);
  bf16_t* emb_p  = (bf16_t*)carve((size_t)E * DV * 2);
  bf16_t* emb_c  = (bf16_t*)carve((size_t)E * DV * 2);
  float*  sums_p = (float*)carve((size_t)N * DV * 4);
  float*  sums_c = (float*)carve((size_t)N * DV * 4);
  float*  deg_p  = (float*)carve((size_t)N * 4);
  float*  deg_c  = (float*)carve((size_t)N * 4);

  auto cvt = [&](const float* s, bf16_t* d, int n) {
    k_cvt_bf16<<<dim3((n + 255) / 256), dim3(256), 0, stream>>>(s, d, n);
  };
  auto zero = [&](float* p, int n) {
    k_zero_f32<<<dim3((n + 255) / 256), dim3(256), 0, stream>>>(p, n);
  };

  cvt(Pv.w1, v_w1, DH * DV);     cvt(Pv.w2, v_w2, DV * DH);
  cvt(Pe.w2, e_w2, DV * DH);
  cvt(Pp.w1, p_w1, DH * 3 * DV); cvt(Pp.w2, p_w2, DV * DH);
  cvt(Pc.w1, c_w1, DH * 3 * DV); cvt(Pc.w2, c_w2, DV * DH);
  cvt(Pa.w1, a_w1, DH * 3 * DV); cvt(Pa.w2, a_w2, DV * DH);

  MlpDev dPv{v_w1, Pv.b1, Pv.g1, Pv.be1, v_w2, Pv.b2, Pv.g2, Pv.be2};
  MlpDev dPe{nullptr, Pe.b1, Pe.g1, Pe.be1, e_w2, Pe.b2, Pe.g2, Pe.be2};
  MlpDev dPp{p_w1, Pp.b1, Pp.g1, Pp.be1, p_w2, Pp.b2, Pp.g2, Pp.be2};
  MlpDev dPc{c_w1, Pc.b1, Pc.g1, Pc.be1, c_w2, Pc.b2, Pc.g2, Pc.be2};
  MlpDev dPa{a_w1, Pa.b1, Pa.g1, Pa.be1, a_w2, Pa.b2, Pa.g2, Pa.be2};

  zero(deg_p, N); zero(deg_c, N);
  k_degree<<<dim3((E + 255) / 256), dim3(256), 0, stream>>>(p_node, deg_p, E);
  k_degree<<<dim3((E + 255) / 256), dim3(256), 0, stream>>>(c_node, deg_c, E);

  const dim3 blk(32);
  const dim3 gN((N + 15) / 16), gE((E + 15) / 16);

  k_node_mlp<<<gN, blk, 0, stream>>>(batch, N, dPv, hidden, hbf);
  k_edge_emb<<<gE, blk, 0, stream>>>(p_ind, E, Pe.w1, Pe.b1, dPe, emb_p);
  k_edge_emb<<<gE, blk, 0, stream>>>(c_ind, E, Pe.w1, Pe.b1, dPe, emb_c);

  for (int it = 0; it < 2; ++it) {            // num_hops - 1 = 2
    zero(sums_p, N * DV); zero(sums_c, N * DV);
    // s_p = MLP_p([h_c, h_p, edge_p]) scattered by p_node
    k_edge_msg<<<gE, blk, 0, stream>>>(hbf, emb_p, p_node, c_node, E, dPp, sums_p);
    // s_c = MLP_c([h_p, h_c, edge_c]) scattered by c_node
    k_edge_msg<<<gE, blk, 0, stream>>>(hbf, emb_c, c_node, p_node, E, dPc, sums_c);
    float* hout = (it == 1) ? (float*)d_out : hidden;
    k_node_aggr<<<gN, blk, 0, stream>>>(hidden, hbf, sums_p, sums_c, deg_p, deg_c,
                                        pm, cm, stok, etok, N, dPa, hout);
  }
}
